// SimplifiedMambaModel_19885698581140
// MI455X (gfx1250) — compile-verified, matrix operand order checked by reading, not run on hardware
//
#include <hip/hip_runtime.h>
#include <math.h>

// ---- model dims ----
#define BB 32
#define TT 2048
#define DD 512
#define SS 64
#define LL 4
#define CC 2
#define MM (BB*TT)        // 65536 rows

typedef __attribute__((ext_vector_type(16))) _Float16 v16h;
typedef __attribute__((ext_vector_type(8)))  float    v8f;
typedef __attribute__((ext_vector_type(4)))  int      v4i;

#define AS1 __attribute__((address_space(1)))
#define AS3 __attribute__((address_space(3)))

__device__ __forceinline__ v8f wmma16(v16h a, v16h b, v8f c) {
  return __builtin_amdgcn_wmma_f32_16x16x32_f16(false, a, false, b, (short)0, c, false, false);
}

// ---- async global->LDS copy (CDNA5), with synchronous fallback ----
#if __has_builtin(__builtin_amdgcn_global_load_async_to_lds_b128)
#define ASYNC_LDS 1
__device__ __forceinline__ void async_b128(const void* g, void* l) {
  void* gv = (void*)g;                       // drop const
  AS1 void* g1 = (AS1 void*)gv;              // to global address space
  AS3 void* l3 = (AS3 void*)l;               // to LDS address space
  __builtin_amdgcn_global_load_async_to_lds_b128(
      (AS1 v4i*)g1, (AS3 v4i*)l3, 0, 0);
}
__device__ __forceinline__ void async_wait0() {
#if __has_builtin(__builtin_amdgcn_s_wait_asynccnt)
  __builtin_amdgcn_s_wait_asynccnt(0);
#else
  asm volatile("s_wait_asynccnt 0" ::: "memory");
#endif
}
#else
__device__ __forceinline__ void async_b128(const void* g, void* l) {
  *(int4*)l = *(const int4*)g;      // synchronous staging fallback
}
__device__ __forceinline__ void async_wait0() {}
#endif

// ---------------------------------------------------------------- embed
__global__ void embed_kernel(const int* __restrict__ tokens,
                             const float* __restrict__ emb,
                             float* __restrict__ x) {
  size_t i = (size_t)blockIdx.x * 256 + threadIdx.x;   // over MM*DD/4
  int row = (int)(i >> 7);          // DD/4 = 128
  int d4  = (int)(i & 127);
  int tok = tokens[row];
  float4 e = ((const float4*)emb)[(size_t)tok * 128 + d4];
  const float sc = 22.627416997969522f;               // sqrt(512)
  float4 o; o.x = e.x*sc; o.y = e.y*sc; o.z = e.z*sc; o.w = e.w*sc;
  ((float4*)x)[i] = o;
}

// ------------------------------------------------------ weight conversion (per layer)
__global__ void conv_w_kernel(const float* __restrict__ Wsp, const float* __restrict__ Wip,
                              const float* __restrict__ bsp, const float* __restrict__ bip,
                              const float* __restrict__ Wg,  const float* __restrict__ Wop,
                              const float* __restrict__ A,
                              _Float16* __restrict__ wc, float* __restrict__ bc,
                              _Float16* __restrict__ wg, _Float16* __restrict__ wop,
                              _Float16* __restrict__ a16) {
  int i = blockIdx.x * 256 + threadIdx.x;              // grid covers DD*DD
  if (i < SS*DD) wc[i]  = (_Float16)(Wsp[i] + Wip[i]);
  if (i < SS)    bc[i]  = bsp[i] + bip[i];
  if (i < DD*DD) wg[i]  = (_Float16)Wg[i];
  if (i < DD*SS) wop[i] = (_Float16)Wop[i];
  if (i < SS*SS) a16[i] = (_Float16)A[i];
}

// ---------------------------------------------------------------- layernorm -> f16
__global__ void ln_kernel(const float* __restrict__ x,
                          const float* __restrict__ w, const float* __restrict__ b,
                          _Float16* __restrict__ xn) {
  int wave = threadIdx.x >> 5, lane = threadIdx.x & 31;
  size_t row = (size_t)blockIdx.x * 8 + wave;
  const float* xr = x + row * DD;
  float v[16], s = 0.f, q = 0.f;
#pragma unroll
  for (int j = 0; j < 16; ++j) { float t = xr[lane + j*32]; v[j] = t; s += t; q += t*t; }
#pragma unroll
  for (int off = 16; off; off >>= 1) { s += __shfl_xor(s, off, 32); q += __shfl_xor(q, off, 32); }
  float m = s * (1.f/DD);
  float r = rsqrtf(q * (1.f/DD) - m*m + 1e-5f);
  _Float16* o = xn + row * DD;
#pragma unroll
  for (int j = 0; j < 16; ++j) {
    int d = lane + j*32;
    o[d] = (_Float16)((v[j] - m) * r * w[d] + b[d]);
  }
}

// ---------------------------------------------------------------- block GEMM core
// Block = 256 threads (8 waves), tile 128(M) x 64(N). The shared 64xK B tile is
// staged through LDS with double-buffered async global->LDS copies; each wave
// runs four 16x16 WMMA accumulators fed by v16h fragments (A direct, B from LDS).
__device__ __forceinline__ void block_gemm64(const _Float16* __restrict__ A, int lda, int mrow0,
                                             const _Float16* __restrict__ Bw, int ldb, int nrow0,
                                             int K, _Float16* ldsB /* [2][64*32] */,
                                             v8f acc[4]) {
  const int tid  = threadIdx.x;
  const int lane = tid & 31;
  const int rsel = lane & 15;
  const int ksel = (lane >> 4) << 4;
  // staging: thread stages one b128 (8 halves) of the 64x32 tile
  const int srow = tid >> 2;              // 0..63
  const int sseg = (tid & 3) << 3;        // 0,8,16,24 halves
  const _Float16* gB = Bw + (size_t)(nrow0 + srow) * ldb + sseg;
  const int soff = srow * 32 + sseg;

  async_b128(gB, ldsB + soff);            // prologue: k0 = 0 into buf 0
  async_wait0();
  __syncthreads();

  const _Float16* ap = A + (size_t)(mrow0 + rsel) * lda + ksel;
  int buf = 0;
  for (int k0 = 0; k0 < K; k0 += 32) {
    if (k0 + 32 < K)
      async_b128(gB + k0 + 32, ldsB + (buf ^ 1) * 2048 + soff);
    v16h af = *(const v16h*)(ap + k0);
    if (k0 + 32 < K) __builtin_prefetch(ap + k0 + 32, 0, 3);
    const _Float16* lb = ldsB + buf * 2048;
#pragma unroll
    for (int n = 0; n < 4; ++n) {
      v16h bf = *(const v16h*)(lb + (n*16 + rsel) * 32 + ksel);
      acc[n] = wmma16(af, bf, acc[n]);
    }
    async_wait0();
    __syncthreads();
    buf ^= 1;
  }
}

// ---- u = xn @ Wc^T + bc  -> stored transposed [T,B,S] for the scan ----
__global__ void gemm_u_kernel(const _Float16* __restrict__ xn,
                              const _Float16* __restrict__ wc,
                              const float* __restrict__ bc,
                              float* __restrict__ u) {
  __shared__ __align__(32) _Float16 ldsB[2 * 64 * 32];
  int wave = threadIdx.x >> 5, lane = threadIdx.x & 31;
  int mrow0 = blockIdx.x * 128 + wave * 16;
  v8f acc[4] = {};
  block_gemm64(xn, DD, mrow0, wc, DD, 0, DD, ldsB, acc);
#pragma unroll
  for (int n = 0; n < 4; ++n) {
    int col = n*16 + (lane & 15);
    float bias = bc[col];
#pragma unroll
    for (int r = 0; r < 8; ++r) {
      int m = mrow0 + r + ((lane >> 4) << 3);
      int b = m >> 11, t = m & (TT - 1);
      u[(size_t)((t << 5) + b) * SS + col] = acc[n][r] + bias;
    }
  }
}

// ---- g = sigmoid(xn @ Wg^T + bg) -> f16 [M,D] ----
__global__ void gemm_g_kernel(const _Float16* __restrict__ xn,
                              const _Float16* __restrict__ wg,
                              const float* __restrict__ bg,
                              _Float16* __restrict__ g) {
  __shared__ __align__(32) _Float16 ldsB[2 * 64 * 32];
  int wave = threadIdx.x >> 5, lane = threadIdx.x & 31;
  int mrow0 = blockIdx.x * 128 + wave * 16;
  int nbase = blockIdx.y * 64;
  v8f acc[4] = {};
  block_gemm64(xn, DD, mrow0, wg, DD, nbase, DD, ldsB, acc);
#pragma unroll
  for (int n = 0; n < 4; ++n) {
    int col = nbase + n*16 + (lane & 15);
    float bias = bg[col];
#pragma unroll
    for (int r = 0; r < 8; ++r) {
      int m = mrow0 + r + ((lane >> 4) << 3);
      float v = acc[n][r] + bias;
      g[(size_t)m * DD + col] = (_Float16)(1.f / (1.f + __expf(-v)));
    }
  }
}

// ---- sequential scan: H = tanh(H @ A^T + U_t), one workgroup, WMMA per step ----
// A kept in registers (B-operand fragments), H double-buffered in LDS, and the
// next step's u-slab (8 KB, contiguous thanks to the [T,B,S] layout) is
// prefetched into LDS with async copies overlapped with the WMMA+tanh work.
__global__ void scan_kernel(const _Float16* __restrict__ a16,
                            const float* __restrict__ u,
                            _Float16* __restrict__ hsq) {
  __shared__ __align__(32) _Float16 H[2][BB][SS];      // 8 KB state, double buffered
  __shared__ __align__(16) float U[2][BB * SS];        // 16 KB u-slab, double buffered
  int wave = threadIdx.x >> 5, lane = threadIdx.x & 31;
  int wm = wave & 1, wn = wave >> 1;                   // 2 M-tiles x 4 N-tiles
  const int rsel = lane & 15, ksel = (lane >> 4) << 4;
  // preload A (B-operand) fragments into registers: rows n, K-major
  v16h bf0 = *(const v16h*)(a16 + (size_t)(wn*16 + rsel) * SS + ksel);
  v16h bf1 = *(const v16h*)(a16 + (size_t)(wn*16 + rsel) * SS + 32 + ksel);
  for (int i = threadIdx.x; i < BB*SS; i += 256) ((_Float16*)H[0])[i] = (_Float16)0.f;
  // prologue: stage u slab for t = 0 (each thread: 2 x b128 = 8 floats)
  {
    const float* g0 = u + threadIdx.x * 8;
    async_b128(g0,     &U[0][threadIdx.x * 8]);
    async_b128(g0 + 4, &U[0][threadIdx.x * 8 + 4]);
    async_wait0();
  }
  __syncthreads();
  int cur = 0;
  const int ar = wm*16 + rsel;
  const int scol = wn*16 + rsel;
  for (int t = 0; t < TT; ++t) {
    if (t + 1 < TT) {                                   // prefetch next slab
      const float* gn = u + ((size_t)(t + 1) << 11) + threadIdx.x * 8;
      async_b128(gn,     &U[cur ^ 1][threadIdx.x * 8]);
      async_b128(gn + 4, &U[cur ^ 1][threadIdx.x * 8 + 4]);
    }
    v16h af0 = *(const v16h*)&H[cur][ar][ksel];
    v16h af1 = *(const v16h*)&H[cur][ar][32 + ksel];
    v8f acc = {};
    acc = wmma16(af0, bf0, acc);
    acc = wmma16(af1, bf1, acc);
#pragma unroll
    for (int r = 0; r < 8; ++r) {
      int b = wm*16 + r + ((lane >> 4) << 3);
      float v = acc[r] + U[cur][b * SS + scol];
      _Float16 hv = (_Float16)tanhf(v);
      H[cur ^ 1][b][scol] = hv;
      hsq[((size_t)b * TT + t) * SS + scol] = hv;
    }
    async_wait0();
    __syncthreads();
    cur ^= 1;
  }
}

// ---- x = (hs @ Wop^T + bop) * g + x  (in place) ----
__global__ void gemm_out_kernel(const _Float16* __restrict__ hsq,
                                const _Float16* __restrict__ wop,
                                const float* __restrict__ bop,
                                const _Float16* __restrict__ g,
                                float* __restrict__ x) {
  __shared__ __align__(32) _Float16 ldsB[2 * 64 * 32];
  int wave = threadIdx.x >> 5, lane = threadIdx.x & 31;
  int mrow0 = blockIdx.x * 128 + wave * 16;
  int nbase = blockIdx.y * 64;
  v8f acc[4] = {};
  block_gemm64(hsq, SS, mrow0, wop, SS, nbase, SS, ldsB, acc);
#pragma unroll
  for (int n = 0; n < 4; ++n) {
    int col = nbase + n*16 + (lane & 15);
    float bias = bop[col];
#pragma unroll
    for (int r = 0; r < 8; ++r) {
      int m = mrow0 + r + ((lane >> 4) << 3);
      size_t idx = (size_t)m * DD + col;
      float v = acc[n][r] + bias;
      x[idx] = v * (float)g[idx] + x[idx];
    }
  }
}

// ---------------------------------------------------------------- final LN + mean over T
__global__ void zero_kernel(float* __restrict__ p, int n) {
  int i = blockIdx.x * 256 + threadIdx.x;
  if (i < n) p[i] = 0.f;
}

__global__ void final_mean_kernel(const float* __restrict__ x,
                                  const float* __restrict__ fw, const float* __restrict__ fb,
                                  float* __restrict__ y) {
  int bb = blockIdx.x >> 3, tc = blockIdx.x & 7;       // grid = B*8
  int wave = threadIdx.x >> 5, lane = threadIdx.x & 31;
  __shared__ float acc[8][DD];
  for (int d = threadIdx.x; d < 8*DD; d += 256) ((float*)acc)[d] = 0.f;
  __syncthreads();
  for (int i = 0; i < 32; ++i) {
    int t = tc*256 + wave*32 + i;
    const float* row = x + ((size_t)bb * TT + t) * DD;
    float v[16], s = 0.f, q = 0.f;
#pragma unroll
    for (int j = 0; j < 16; ++j) { float xv = row[lane + j*32]; v[j] = xv; s += xv; q += xv*xv; }
#pragma unroll
    for (int off = 16; off; off >>= 1) { s += __shfl_xor(s, off, 32); q += __shfl_xor(q, off, 32); }
    float m = s * (1.f/DD);
    float r = rsqrtf(q * (1.f/DD) - m*m + 1e-5f);
#pragma unroll
    for (int j = 0; j < 16; ++j) {
      int d = lane + j*32;
      acc[wave][d] += (v[j] - m) * r * fw[d] + fb[d];
    }
  }
  __syncthreads();
  for (int d = threadIdx.x; d < DD; d += 256) {
    float s = 0.f;
#pragma unroll
    for (int w = 0; w < 8; ++w) s += acc[w][d];
    atomicAdd(&y[bb*DD + d], s * (1.f/TT));
  }
}

// ---------------------------------------------------------------- tiny head
__global__ void head_kernel(const float* __restrict__ y,
                            const float* __restrict__ W1, const float* __restrict__ b1,
                            const float* __restrict__ W2, const float* __restrict__ b2,
                            float* __restrict__ out) {
  int b = blockIdx.x, j = threadIdx.x;
  __shared__ float ys[DD];
  __shared__ float hh[DD/2];
  ys[j] = y[b*DD + j]; ys[j + 256] = y[b*DD + 256 + j];
  __syncthreads();
  float s = b1[j];
  for (int d = 0; d < DD; ++d) s += ys[d] * W1[j*DD + d];
  hh[j] = fmaxf(s, 0.f);
  __syncthreads();
  if (j < CC) {
    float o = b2[j];
    for (int d = 0; d < DD/2; ++d) o += hh[d] * W2[j*(DD/2) + d];
    out[b*CC + j] = o;
  }
}

// ---------------------------------------------------------------- launch
extern "C" void kernel_launch(void* const* d_in, const int* in_sizes, int n_in,
                              void* d_out, int out_size, void* d_ws, size_t ws_size,
                              hipStream_t stream) {
  (void)in_sizes; (void)n_in; (void)out_size; (void)ws_size;
  const int*   tokens = (const int*)d_in[0];
  const float* emb  = (const float*)d_in[1];
  const float* ln_w = (const float*)d_in[2];
  const float* ln_b = (const float*)d_in[3];
  const float* Wsp  = (const float*)d_in[4];
  const float* bsp  = (const float*)d_in[5];
  const float* Wip  = (const float*)d_in[6];
  const float* bip  = (const float*)d_in[7];
  const float* Wop  = (const float*)d_in[8];
  const float* bop  = (const float*)d_in[9];
  const float* Wg   = (const float*)d_in[10];
  const float* bg   = (const float*)d_in[11];
  const float* Aw   = (const float*)d_in[12];
  const float* fw   = (const float*)d_in[13];
  const float* fb   = (const float*)d_in[14];
  const float* W1   = (const float*)d_in[15];
  const float* b1   = (const float*)d_in[16];
  const float* W2   = (const float*)d_in[17];
  const float* b2   = (const float*)d_in[18];
  float* out = (float*)d_out;

  char* p = (char*)d_ws;
  auto alloc = [&](size_t bytes) -> char* {
    char* r = p; p += (bytes + 255) & ~(size_t)255; return r;
  };
  float*    x   = (float*)   alloc((size_t)MM * DD * 4);   // 128 MB
  _Float16* xn  = (_Float16*)alloc((size_t)MM * DD * 2);   //  64 MB
  float*    u   = (float*)   alloc((size_t)MM * SS * 4);   //  16 MB  [T,B,S]
  _Float16* g   = (_Float16*)alloc((size_t)MM * DD * 2);   //  64 MB
  _Float16* hsq = (_Float16*)alloc((size_t)MM * SS * 2);   //   8 MB
  _Float16* wc  = (_Float16*)alloc((size_t)SS * DD * 2);
  float*    bc  = (float*)   alloc((size_t)SS * 4);
  _Float16* wgh = (_Float16*)alloc((size_t)DD * DD * 2);
  _Float16* wph = (_Float16*)alloc((size_t)DD * SS * 2);
  _Float16* a16 = (_Float16*)alloc((size_t)SS * SS * 2);
  float*    y   = (float*)   alloc((size_t)BB * DD * 4);

  dim3 blk(256);
  embed_kernel<<<MM * (DD/4) / 256, blk, 0, stream>>>(tokens, emb, x);

  for (int l = 0; l < LL; ++l) {
    conv_w_kernel<<<DD*DD/256, blk, 0, stream>>>(
        Wsp + (size_t)l*SS*DD, Wip + (size_t)l*SS*DD,
        bsp + l*SS, bip + l*SS,
        Wg + (size_t)l*DD*DD, Wop + (size_t)l*DD*SS, Aw + (size_t)l*SS*SS,
        wc, bc, wgh, wph, a16);
    ln_kernel<<<MM/8, blk, 0, stream>>>(x, ln_w + l*DD, ln_b + l*DD, xn);
    gemm_u_kernel<<<MM/128, blk, 0, stream>>>(xn, wc, bc, u);
    gemm_g_kernel<<<dim3(MM/128, DD/64), blk, 0, stream>>>(xn, wgh, bg + l*DD, g);
    scan_kernel<<<1, blk, 0, stream>>>(a16, u, hsq);
    gemm_out_kernel<<<dim3(MM/128, DD/64), blk, 0, stream>>>(hsq, wph, bop + l*DD, g, x);
  }

  zero_kernel<<<(BB*DD)/256, blk, 0, stream>>>(y, BB*DD);
  final_mean_kernel<<<BB*8, blk, 0, stream>>>(x, fw, fb, y);
  head_kernel<<<BB, blk, 0, stream>>>(y, W1, b1, W2, b2, out);
}
